// HOPE_7275674599449
// MI455X (gfx1250) — matrix operand order repaired and checked by reading, not compiled, not run on hardware
//
#include <hip/hip_runtime.h>

// ---------------------------------------------------------------------------
// HOPE block forward for MI455X (gfx1250, wave32, WMMA bf16 16x16x32, f32 acc)
// ---------------------------------------------------------------------------

typedef __attribute__((ext_vector_type(16))) __bf16 v16bf;
typedef __attribute__((ext_vector_type(8)))  float  v8f;

union FragB16 { v16bf v; uint4 q[2]; };

#define TM 128
#define TN 128
#define TK 32
#define NTHREADS 256
#define LDSS 48   // LDS row stride in bf16 elems (96B: 16B aligned, conflict-skewed)

__device__ __forceinline__ unsigned short f2bf(float f) {
    unsigned int u = __float_as_uint(f);
    unsigned int r = u + 0x7FFFu + ((u >> 16) & 1u);   // round-to-nearest-even
    return (unsigned short)(r >> 16);
}

// activation: 0 = none, 1 = elu(x)+1, 2 = exact gelu
template <int ACT>
__device__ __forceinline__ float epilogue_act(float x) {
    if (ACT == 1) return (x > 0.0f) ? (x + 1.0f) : __expf(x);
    if (ACT == 2) return 0.5f * x * (1.0f + erff(x * 0.70710678118654752f));
    return x;
}

// C[M,N] = act(A[M,K] @ B[K,N] + bias[N])   (B row-major KxN; all dims 128/32 multiples)
template <int ACT>
__global__ __launch_bounds__(NTHREADS) void gemm_kernel(
    const float* __restrict__ A, int lda, long long aBatch,
    const float* __restrict__ Bm, int ldb, long long bBatch,
    const float* __restrict__ bias,
    float* __restrict__ C, int ldc, long long cBatch, int K)
{
    __shared__ unsigned short As[TM][LDSS];
    __shared__ unsigned short Bs[TN][LDSS];   // stored transposed: Bs[n][k]

    const int t  = threadIdx.x;
    const int m0 = blockIdx.y * TM;
    const int n0 = blockIdx.x * TN;
    A  += (long long)blockIdx.z * aBatch;
    Bm += (long long)blockIdx.z * bBatch;
    C  += (long long)blockIdx.z * cBatch;

    const int lane = t & 31, wave = t >> 5;
    const int wm = wave & 3, wn = wave >> 2;       // 4 waves along M, 2 along N
    const int lrow = lane & 15, hsel = lane >> 4;  // half-wave selector

    v8f acc[2][4];
    const v8f vzero = {0.f,0.f,0.f,0.f,0.f,0.f,0.f,0.f};
#pragma unroll
    for (int i = 0; i < 2; ++i)
#pragma unroll
        for (int j = 0; j < 4; ++j) acc[i][j] = vzero;

    for (int k0 = 0; k0 < K; k0 += TK) {
        __syncthreads();
        {   // stage A tile 128x32 fp32 -> bf16 LDS; thread: row t/2, 16-col half
            const int r  = t >> 1;
            const int c0 = (t & 1) * 16;
            const float* ga = A + (long long)(m0 + r) * lda + k0 + c0;
#pragma unroll
            for (int i = 0; i < 4; ++i) {
                float4 f = *(const float4*)(ga + i * 4);
                uint2 p;
                p.x = (unsigned)f2bf(f.x) | ((unsigned)f2bf(f.y) << 16);
                p.y = (unsigned)f2bf(f.z) | ((unsigned)f2bf(f.w) << 16);
                *(uint2*)&As[r][c0 + i * 4] = p;
            }
        }
        {   // stage B tile 32x128 fp32 -> bf16 LDS transposed (Bs[n][k])
            const int kr = (t >> 5) << 2;     // 0..28
            const int nc = (t & 31) << 2;     // 0..124
            float tmp[4][4];
#pragma unroll
            for (int j = 0; j < 4; ++j) {
                float4 f = *(const float4*)(Bm + (long long)(k0 + kr + j) * ldb + n0 + nc);
                tmp[j][0] = f.x; tmp[j][1] = f.y; tmp[j][2] = f.z; tmp[j][3] = f.w;
            }
#pragma unroll
            for (int c = 0; c < 4; ++c) {
                uint2 p;
                p.x = (unsigned)f2bf(tmp[0][c]) | ((unsigned)f2bf(tmp[1][c]) << 16);
                p.y = (unsigned)f2bf(tmp[2][c]) | ((unsigned)f2bf(tmp[3][c]) << 16);
                *(uint2*)&Bs[nc + c][kr] = p;
            }
        }
        __syncthreads();

        // fragments per ISA 16-bit layouts
        FragB16 af[2], bf[4];
#pragma unroll
        for (int tm = 0; tm < 2; ++tm) {
            const int row = wm * 32 + tm * 16 + lrow;
            const int kl  = hsel * 8;              // lanes 0-15: K0..7/16..23; 16-31: K8..15/24..31
            af[tm].q[0] = *(const uint4*)&As[row][kl];
            af[tm].q[1] = *(const uint4*)&As[row][kl + 16];
        }
#pragma unroll
        for (int tn = 0; tn < 4; ++tn) {
            const int n  = wn * 64 + tn * 16 + lrow;
            const int kl = hsel * 8;               // lanes 0-15: K0..7/16..23; 16-31: K8..15/24..31
            bf[tn].q[0] = *(const uint4*)&Bs[n][kl];
            bf[tn].q[1] = *(const uint4*)&Bs[n][kl + 16];
        }
#pragma unroll
        for (int tm = 0; tm < 2; ++tm)
#pragma unroll
            for (int tn = 0; tn < 4; ++tn)
                acc[tm][tn] = __builtin_amdgcn_wmma_f32_16x16x32_bf16(
                    false, af[tm].v, false, bf[tn].v, (short)0, acc[tm][tn], false, false);
    }

    // epilogue: C/D layout — VGPR j: lanes0-15 M=j, lanes16-31 M=j+8; N = lane&15
#pragma unroll
    for (int tn = 0; tn < 4; ++tn) {
        const int col = n0 + wn * 64 + tn * 16 + lrow;
        const float bb = bias ? bias[col] : 0.0f;
#pragma unroll
        for (int tm = 0; tm < 2; ++tm) {
            const int rbase = m0 + wm * 32 + tm * 16 + hsel * 8;
#pragma unroll
            for (int j = 0; j < 8; ++j) {
                float vl = epilogue_act<ACT>(acc[tm][tn][j] + bb);
                C[(long long)(rbase + j) * ldc + col] = vl;
            }
        }
    }
}

// scores[b,t,s] = (q_t . k_s) * decay^(t-1-s), strictly causal (s<t), else 0.
// q,k: [B,S,D] row-major.  Same WMMA tiling; B side (k) is N-major so it stages like A.
__global__ __launch_bounds__(NTHREADS) void scores_kernel(
    const float* __restrict__ Q, const float* __restrict__ Km,
    const float* __restrict__ decay_param,
    float* __restrict__ C, int S, int D)
{
    const int t  = threadIdx.x;
    const int m0 = blockIdx.y * TM;   // query rows
    const int n0 = blockIdx.x * TN;   // key cols
    Q  += (long long)blockIdx.z * S * D;
    Km += (long long)blockIdx.z * S * D;
    C  += (long long)blockIdx.z * S * S;

    if (n0 >= m0 + TM - 1) {          // whole tile strictly masked -> zero-fill
        const float4 z = {0.f, 0.f, 0.f, 0.f};
        for (int i = t; i < TM * TN / 4; i += NTHREADS) {
            int r = i >> 5, c = (i & 31) << 2;
            *(float4*)&C[(long long)(m0 + r) * S + n0 + c] = z;
        }
        return;
    }

    __shared__ unsigned short As[TM][LDSS];
    __shared__ unsigned short Bs[TN][LDSS];

    const int lane = t & 31, wave = t >> 5;
    const int wm = wave & 3, wn = wave >> 2;
    const int lrow = lane & 15, hsel = lane >> 4;

    v8f acc[2][4];
    const v8f vzero = {0.f,0.f,0.f,0.f,0.f,0.f,0.f,0.f};
#pragma unroll
    for (int i = 0; i < 2; ++i)
#pragma unroll
        for (int j = 0; j < 4; ++j) acc[i][j] = vzero;

    for (int k0 = 0; k0 < D; k0 += TK) {
        __syncthreads();
        {
            const int r  = t >> 1;
            const int c0 = (t & 1) * 16;
            const float* gq = Q  + (long long)(m0 + r) * D + k0 + c0;
            const float* gk = Km + (long long)(n0 + r) * D + k0 + c0;
#pragma unroll
            for (int i = 0; i < 4; ++i) {
                float4 f = *(const float4*)(gq + i * 4);
                uint2 p;
                p.x = (unsigned)f2bf(f.x) | ((unsigned)f2bf(f.y) << 16);
                p.y = (unsigned)f2bf(f.z) | ((unsigned)f2bf(f.w) << 16);
                *(uint2*)&As[r][c0 + i * 4] = p;
                float4 g = *(const float4*)(gk + i * 4);
                uint2 pb;
                pb.x = (unsigned)f2bf(g.x) | ((unsigned)f2bf(g.y) << 16);
                pb.y = (unsigned)f2bf(g.z) | ((unsigned)f2bf(g.w) << 16);
                *(uint2*)&Bs[r][c0 + i * 4] = pb;
            }
        }
        __syncthreads();

        FragB16 af[2], bf[4];
#pragma unroll
        for (int tm = 0; tm < 2; ++tm) {
            const int row = wm * 32 + tm * 16 + lrow;
            const int kl  = hsel * 8;
            af[tm].q[0] = *(const uint4*)&As[row][kl];
            af[tm].q[1] = *(const uint4*)&As[row][kl + 16];
        }
#pragma unroll
        for (int tn = 0; tn < 4; ++tn) {
            const int n  = wn * 64 + tn * 16 + lrow;
            const int kl = hsel * 8;
            bf[tn].q[0] = *(const uint4*)&Bs[n][kl];
            bf[tn].q[1] = *(const uint4*)&Bs[n][kl + 16];
        }
#pragma unroll
        for (int tm = 0; tm < 2; ++tm)
#pragma unroll
            for (int tn = 0; tn < 4; ++tn)
                acc[tm][tn] = __builtin_amdgcn_wmma_f32_16x16x32_bf16(
                    false, af[tm].v, false, bf[tn].v, (short)0, acc[tm][tn], false, false);
    }

    const float p   = decay_param[0];
    const float dec = 1.0f / (1.0f + __expf(-p));
    const float l2d = __log2f(dec);

#pragma unroll
    for (int tn = 0; tn < 4; ++tn) {
        const int s = n0 + wn * 64 + tn * 16 + lrow;
#pragma unroll
        for (int tm = 0; tm < 2; ++tm) {
            const int rbase = m0 + wm * 32 + tm * 16 + hsel * 8;
#pragma unroll
            for (int j = 0; j < 8; ++j) {
                const int tq   = rbase + j;
                const int expo = tq - 1 - s;
                float w = (expo < 0) ? 0.0f
                        : (expo == 0 ? 1.0f : exp2f((float)expo * l2d));
                C[(long long)tq * S + s] = acc[tm][tn][j] * w;
            }
        }
    }
}

// out[row] = LayerNorm(a[row] + b[row]) * g + beta   (D = 1024, one block per row)
__global__ __launch_bounds__(NTHREADS) void ln_residual(
    const float* __restrict__ a, const float* __restrict__ b,
    const float* __restrict__ g, const float* __restrict__ be,
    float* __restrict__ out, int D)
{
    const int row = blockIdx.x, t = threadIdx.x;
    const float* pa = a + (long long)row * D;
    const float* pb = b + (long long)row * D;
    float vals[4], s = 0.f, sq = 0.f;
#pragma unroll
    for (int i = 0; i < 4; ++i) {
        int idx = t + i * NTHREADS;
        float x = pa[idx] + pb[idx];
        vals[i] = x; s += x; sq += x * x;
    }
    __shared__ float rs[NTHREADS], rq[NTHREADS];
    rs[t] = s; rq[t] = sq; __syncthreads();
    for (int off = NTHREADS / 2; off > 0; off >>= 1) {
        if (t < off) { rs[t] += rs[t + off]; rq[t] += rq[t + off]; }
        __syncthreads();
    }
    const float mean = rs[0] / (float)D;
    const float var  = rq[0] / (float)D - mean * mean;
    const float inv  = rsqrtf(var + 1e-5f);
    float* po = out + (long long)row * D;
#pragma unroll
    for (int i = 0; i < 4; ++i) {
        int idx = t + i * NTHREADS;
        po[idx] = (vals[i] - mean) * inv * g[idx] + be[idx];
    }
}

// ---------------------------------------------------------------------------

extern "C" void kernel_launch(void* const* d_in, const int* in_sizes, int n_in,
                              void* d_out, int out_size, void* d_ws, size_t ws_size,
                              hipStream_t stream) {
    const int Bb = 2, S = 2048, D = 1024, L = 4, V = 32000, E = 4 * D; // 4096
    const int M  = Bb * S;  // 4096

    const float* x   = (const float*)d_in[0];
    const float* Wq  = (const float*)d_in[1];  const float* bq = (const float*)d_in[2];
    const float* Wk  = (const float*)d_in[3];  const float* bk = (const float*)d_in[4];
    const float* Wv  = (const float*)d_in[5];  const float* bv = (const float*)d_in[6];
    const float* Wo  = (const float*)d_in[7];  const float* bo = (const float*)d_in[8];
    const float* dp  = (const float*)d_in[9];
    const float* gf  = (const float*)d_in[10]; const float* bf_ = (const float*)d_in[11];
    const float* W1  = (const float*)d_in[12]; const float* b1 = (const float*)d_in[13];
    const float* W2  = (const float*)d_in[14]; const float* b2 = (const float*)d_in[15];
    const float* gc  = (const float*)d_in[16]; const float* bc = (const float*)d_in[17];
    const float* Wh  = (const float*)d_in[18]; const float* bh = (const float*)d_in[19];
    float* out = (float*)d_out;

    // workspace: R0,R1,R2 = 4096*1024 floats each; R3 = 4096*4096 floats
    const long long SZ = (long long)M * D;
    float* R0 = (float*)d_ws;
    float* R1 = R0 + SZ;
    float* R2 = R1 + SZ;
    float* R3 = R2 + SZ;                       // 16.78M floats (holds scores or MLP tmp)

    const dim3 blk(NTHREADS);
    const dim3 gD (D / TN, M / TM, 1);         // N=1024 GEMMs on full M
    const dim3 gE (E / TN, M / TM, 1);         // N=4096
    const dim3 gV (V / TN, M / TM, 1);         // N=32000 -> 250 tiles
    const dim3 gS (S / TN, S / TM, Bb);        // scores per batch
    const dim3 gAV(D / TN, S / TM, Bb);        // scores @ v per batch

    // q, k(=elu+1), v projections
    gemm_kernel<0><<<gD, blk, 0, stream>>>(x, D, 0, Wq, D, 0, bq, R0, D, 0, D);
    gemm_kernel<1><<<gD, blk, 0, stream>>>(x, D, 0, Wk, D, 0, bk, R1, D, 0, D);
    gemm_kernel<0><<<gD, blk, 0, stream>>>(x, D, 0, Wv, D, 0, bv, R2, D, 0, D);

    // decay-masked causal scores: [B,S,S] in R3
    scores_kernel<<<gS, blk, 0, stream>>>(R0, R1, dp, R3, S, D);

    // attn = scores @ v  -> R0
    gemm_kernel<0><<<gAV, blk, 0, stream>>>(R3, S, (long long)S * S,
                                            R2, D, (long long)S * D,
                                            nullptr, R0, D, (long long)S * D, S);
    // fast = attn @ Wo + bo -> R1
    gemm_kernel<0><<<gD, blk, 0, stream>>>(R0, D, 0, Wo, D, 0, bo, R1, D, 0, D);
    // h = LN(x + fast) -> R2
    ln_residual<<<dim3(M), blk, 0, stream>>>(x, R1, gf, bf_, R2, D);

    // CMS stack
    for (int l = 0; l < L; ++l) {
        const float* W1l = W1 + (long long)l * D * E;
        const float* b1l = b1 + (long long)l * E;
        const float* W2l = W2 + (long long)l * E * D;
        const float* b2l = b2 + (long long)l * D;
        gemm_kernel<2><<<gE, blk, 0, stream>>>(R2, D, 0, W1l, E, 0, b1l, R3, E, 0, D);
        gemm_kernel<0><<<gD, blk, 0, stream>>>(R3, E, 0, W2l, D, 0, b2l, R0, D, 0, E);
        ln_residual<<<dim3(M), blk, 0, stream>>>(R2, R0, gc + (long long)l * D,
                                                 bc + (long long)l * D, R2, D);
    }

    // LM head -> d_out
    gemm_kernel<0><<<gV, blk, 0, stream>>>(R2, D, 0, Wh, V, 0, bh, out, V, 0, D);
    (void)in_sizes; (void)n_in; (void)out_size; (void)ws_size;
}
